// SparseBayesianLinear_1073741824318
// MI455X (gfx1250) — matrix operand order repaired and checked by reading, not compiled, not run on hardware
//
#include <hip/hip_runtime.h>
#include <hip/hip_bf16.h>
#include <math.h>

// ---------------------------------------------------------------------------
// SparseBayesianLinear fused dual-GEMM for MI455X (gfx1250, wave32, WMMA).
//   scores = (x @ keys^T) * 1/sqrt(D),  keys = mu * softplus(sigma)
//   comp   =  x @ mu^T
//   masked = comp * relu(scores - gate)
//   outputs: [final(=masked) | scores | masked], each 8192x2048 f32
// bf16 prepass -> fused K-loop with v_wmma_f32_16x16x32_bf16.
// Staging uses GLOBAL_LOAD_ASYNC_TO_LDS_B128 (ASYNCcnt) with triple-buffered
// LDS and one workgroup barrier per K-step; outputs use non-temporal stores
// so the 201MB of results don't evict the L2-resident operands.
// ---------------------------------------------------------------------------

typedef __attribute__((ext_vector_type(16))) __bf16 v16bf;
typedef __attribute__((ext_vector_type(4)))  __bf16 v4bf;
typedef __attribute__((ext_vector_type(8)))  float  v8f;

#define N_ROWS 8192          // 4*2048 flattened tokens
#define DIMK   2048          // in_features
#define DIMO   2048          // out_features
#define TILE_N 128           // rows per workgroup
#define TILE_O 128           // cols per workgroup
#define KSTEP  32            // K per WMMA
#define NSTEPS (DIMK / KSTEP)
#define LDS_STRIDE 40        // bf16 per LDS row (32 + 8 pad, keeps 16B align)
#define BUF_ELEMS (TILE_N * LDS_STRIDE)      // 5120 bf16
#define BUF_BYTES (BUF_ELEMS * 2)            // 10240 B

union Frag {
    v16bf bf;
    uint4 u[2];
};

// async global->LDS 16B copy: lds byte offset (VGPR), global byte offset
// (VGPR) added to uniform 64-bit base (SGPR pair). Tracked by ASYNCcnt.
#define ASYNC_CP16(ldsoff, voff, sbase)                                   \
    asm volatile("global_load_async_to_lds_b128 %0, %1, %2"               \
                 :: "v"(ldsoff), "v"(voff), "s"(sbase) : "memory")

#define WG_BARRIER()                                                      \
    asm volatile("s_barrier_signal -1\n\ts_barrier_wait -1" ::: "memory")

// ---------------- prepass: f32 -> bf16 conversions -------------------------

__global__ void cvt_x_kernel(const float* __restrict__ x,
                             __bf16* __restrict__ xb, int n4) {
    int i = blockIdx.x * blockDim.x + threadIdx.x;
    if (i >= n4) return;
    float4 v = ((const float4*)x)[i];
    v4bf o = { (__bf16)v.x, (__bf16)v.y, (__bf16)v.z, (__bf16)v.w };
    ((v4bf*)xb)[i] = o;
}

__device__ __forceinline__ float softplus_f(float s) {
    return (s > 20.0f) ? s : log1pf(expf(s));
}

__global__ void cvt_w_kernel(const float* __restrict__ mu,
                             const float* __restrict__ sg,
                             __bf16* __restrict__ mub,
                             __bf16* __restrict__ keyb, int n4) {
    int i = blockIdx.x * blockDim.x + threadIdx.x;
    if (i >= n4) return;
    float4 m = ((const float4*)mu)[i];
    float4 s = ((const float4*)sg)[i];
    v4bf om = { (__bf16)m.x, (__bf16)m.y, (__bf16)m.z, (__bf16)m.w };
    v4bf ok = { (__bf16)(m.x * softplus_f(s.x)),
                (__bf16)(m.y * softplus_f(s.y)),
                (__bf16)(m.z * softplus_f(s.z)),
                (__bf16)(m.w * softplus_f(s.w)) };
    ((v4bf*)mub)[i] = om;
    ((v4bf*)keyb)[i] = ok;
}

// ---------------- fused dual-GEMM + gating epilogue ------------------------

__global__ void __launch_bounds__(256)
sbl_gemm_kernel(const __bf16* __restrict__ xb,
                const __bf16* __restrict__ mub,
                const __bf16* __restrict__ keyb,
                const float*  __restrict__ gate,
                float* __restrict__ out) {
    __shared__ __bf16 lsX [3][BUF_ELEMS];
    __shared__ __bf16 lsMu[3][BUF_ELEMS];
    __shared__ __bf16 lsKy[3][BUF_ELEMS];

    const int tid  = threadIdx.x;
    const int lane = tid & 31;
    const int wid  = tid >> 5;          // 8 waves
    const int lm   = lane & 15;         // lane within half
    const int lh   = lane >> 4;         // lane half (0/1)

    const int waveM = wid & 1;          // 2 row-wave groups of 64 rows
    const int waveN = wid >> 1;         // 4 col-wave groups of 32 cols

    const int nBlock = blockIdx.y * TILE_N;
    const int oBlock = blockIdx.x * TILE_O;

    // accumulators: 4 M-tiles x 2 N-tiles x 2 matrices
    v8f accMu[4][2];
    v8f accKy[4][2];
#pragma unroll
    for (int m = 0; m < 4; ++m)
#pragma unroll
        for (int t = 0; t < 2; ++t) { accMu[m][t] = (v8f)0.0f; accKy[m][t] = (v8f)0.0f; }

    float gv[2];
#pragma unroll
    for (int t = 0; t < 2; ++t)
        gv[t] = gate[oBlock + waveN * 32 + t * 16 + lm];

    // staging geometry: tile = 128 rows x 32 bf16 = 512 x 16B chunks, 2/thread
    const int c0 = tid, c1 = tid + 256;
    const int r0 = c0 >> 2, kc0 = c0 & 3;
    const int r1 = c1 >> 2, kc1 = c1 & 3;

    // per-thread global byte offsets within a tile (same for x/mu/keys)
    const unsigned gvo0 = (unsigned)(r0 * (DIMK * 2) + kc0 * 16);
    const unsigned gvo1 = (unsigned)(r1 * (DIMK * 2) + kc1 * 16);
    // per-thread LDS byte offsets within one buffer
    const unsigned lvo0 = (unsigned)(r0 * (LDS_STRIDE * 2) + kc0 * 16);
    const unsigned lvo1 = (unsigned)(r1 * (LDS_STRIDE * 2) + kc1 * 16);
    // LDS base byte addresses (wave-relative offsets after truncation)
    const unsigned ldsXb = (unsigned)(uintptr_t)&lsX[0][0];
    const unsigned ldsMb = (unsigned)(uintptr_t)&lsMu[0][0];
    const unsigned ldsKb = (unsigned)(uintptr_t)&lsKy[0][0];

    // uniform 64-bit global bases for this block's tile rows
    const unsigned long long gxBase = (unsigned long long)(uintptr_t)xb
                                    + (unsigned long long)nBlock * (DIMK * 2);
    const unsigned long long gmBase = (unsigned long long)(uintptr_t)mub
                                    + (unsigned long long)oBlock * (DIMK * 2);
    const unsigned long long gkBase = (unsigned long long)(uintptr_t)keyb
                                    + (unsigned long long)oBlock * (DIMK * 2);

    // issue one staging batch (6 async b128 copies) for K-step `step` -> buf
    auto issue_batch = [&](int step, int buf) {
        const unsigned long long kb = (unsigned long long)(step * KSTEP * 2);
        const unsigned long long sx = gxBase + kb;
        const unsigned long long sm = gmBase + kb;
        const unsigned long long sk = gkBase + kb;
        const unsigned lb = (unsigned)(buf * BUF_BYTES);
        ASYNC_CP16(ldsXb + lb + lvo0, gvo0, sx);
        ASYNC_CP16(ldsXb + lb + lvo1, gvo1, sx);
        ASYNC_CP16(ldsMb + lb + lvo0, gvo0, sm);
        ASYNC_CP16(ldsMb + lb + lvo1, gvo1, sm);
        ASYNC_CP16(ldsKb + lb + lvo0, gvo0, sk);
        ASYNC_CP16(ldsKb + lb + lvo1, gvo1, sk);
    };

    issue_batch(0, 0);                 // prologue: batch(0) -> buf 0

    for (int i = 0; i < NSTEPS; ++i) {
        const int buf = i % 3;
        if (i + 1 < NSTEPS) {
            issue_batch(i + 1, (i + 1) % 3);
            // newest batch (6 ops) may stay in flight; older ones must land
            asm volatile("s_wait_asynccnt 6" ::: "memory");
        } else {
            asm volatile("s_wait_asynccnt 0" ::: "memory");
        }
        WG_BARRIER();                  // all waves' buf[i%3] staged

        // ---- A fragments: lane half 0 -> K0..15, half 1 -> K16..31
        Frag fa[4];
#pragma unroll
        for (int m = 0; m < 4; ++m) {
            const __bf16* p = &lsX[buf][(waveM * 64 + m * 16 + lm) * LDS_STRIDE];
            fa[m].u[0] = *(const uint4*)(p + lh * 16);
            fa[m].u[1] = *(const uint4*)(p + lh * 16 + 8);
        }

#pragma unroll
        for (int t = 0; t < 2; ++t) {
            // B: lane half 0 -> K0..15, half 1 -> K16..31; column = lm
            const int orow = (waveN * 32 + t * 16 + lm) * LDS_STRIDE;
            Frag fbm, fbk;
            fbm.u[0] = *(const uint4*)(&lsMu[buf][orow] + lh * 16);
            fbm.u[1] = *(const uint4*)(&lsMu[buf][orow] + lh * 16 + 8);
            fbk.u[0] = *(const uint4*)(&lsKy[buf][orow] + lh * 16);
            fbk.u[1] = *(const uint4*)(&lsKy[buf][orow] + lh * 16 + 8);
#pragma unroll
            for (int m = 0; m < 4; ++m) {
                accMu[m][t] = __builtin_amdgcn_wmma_f32_16x16x32_bf16(
                    false, fa[m].bf, false, fbm.bf, (short)0, accMu[m][t], false, false);
                accKy[m][t] = __builtin_amdgcn_wmma_f32_16x16x32_bf16(
                    false, fa[m].bf, false, fbk.bf, (short)0, accKy[m][t], false, false);
            }
        }
        // no trailing barrier: WAR distance is 2 iters (triple buffer), and all
        // ds_loads above are consumed by WMMA before the next iter's barrier.
    }

    // ---- epilogue: scores, relu gate, mask; non-temporal streaming stores ----
    const float inv_sqrt_d = 0.02209708691207961f;   // 1/sqrt(2048)
    const size_t NO = (size_t)N_ROWS * DIMO;
#pragma unroll
    for (int m = 0; m < 4; ++m) {
#pragma unroll
        for (int t = 0; t < 2; ++t) {
            const int o = oBlock + waveN * 32 + t * 16 + lm;
            const float g = gv[t];
#pragma unroll
            for (int r = 0; r < 8; ++r) {
                const int n = nBlock + waveM * 64 + m * 16 + r + lh * 8;
                const float sc  = accKy[m][t][r] * inv_sqrt_d;
                const float raw = fmaxf(sc - g, 0.0f);
                const float msk = accMu[m][t][r] * raw;
                const size_t idx = (size_t)n * DIMO + o;
                __builtin_nontemporal_store(msk, &out[idx]);           // final
                __builtin_nontemporal_store(sc,  &out[NO + idx]);      // scores
                __builtin_nontemporal_store(msk, &out[2 * NO + idx]);  // masked
            }
        }
    }
}

// ---------------------------------------------------------------------------

extern "C" void kernel_launch(void* const* d_in, const int* in_sizes, int n_in,
                              void* d_out, int out_size, void* d_ws, size_t ws_size,
                              hipStream_t stream) {
    const float* x    = (const float*)d_in[0];
    const float* mu   = (const float*)d_in[1];
    const float* sg   = (const float*)d_in[2];
    const float* gate = (const float*)d_in[3];
    float* out = (float*)d_out;

    // workspace (bf16): x [8192*2048] | mu [2048*2048] | keys [2048*2048]
    __bf16* xb   = (__bf16*)d_ws;
    __bf16* mub  = xb  + (size_t)N_ROWS * DIMK;
    __bf16* keyb = mub + (size_t)DIMO * DIMK;

    {
        int n4 = (N_ROWS * DIMK) / 4;
        cvt_x_kernel<<<n4 / 256, 256, 0, stream>>>(x, xb, n4);
    }
    {
        int n4 = (DIMO * DIMK) / 4;
        cvt_w_kernel<<<n4 / 256, 256, 0, stream>>>(mu, sg, mub, keyb, n4);
    }
    dim3 grid(DIMO / TILE_O, N_ROWS / TILE_N);        // 16 x 64
    sbl_gemm_kernel<<<grid, 256, 0, stream>>>(xb, mub, keyb, gate, out);
}
